// Seg3dLossless_62294205661986
// MI455X (gfx1250) — matrix operand order repaired and checked by reading, not compiled, not run on hardware
//
#include <hip/hip_runtime.h>

// ---------------------------------------------------------------------------
// Seg3dLossless dense-equivalent pipeline for MI455X (gfx1250, wave32, WMMA)
//
// Levels: 17 -> 33 -> 65 -> 129 -> (pure upsample) 257
// MLP 3->64->64->1 evaluated densely per level via v_wmma_f32_16x16x32_f16.
// Weight B-fragments are built once per wave and amortized over a grid-stride
// tile loop; activations re-stripe through a per-wave LDS slab (ds_load_b128).
// Masks/upsample kernels are bandwidth-trivial (all buffers L2-resident);
// the dominant cost is the mandatory 67.9 MB coalesced store of the 257^3
// output (~3 us at 23.3 TB/s).
// ---------------------------------------------------------------------------

typedef _Float16 v16h __attribute__((ext_vector_type(16)));
typedef _Float16 half8 __attribute__((ext_vector_type(8)));
typedef float    v8f  __attribute__((ext_vector_type(8)));

#define WAVES_PER_BLOCK 8
#define HID 64

// --- pack fp32 weights to f16 (K-major, [k*64+n]) and init level-0 accum ----
__global__ __launch_bounds__(256) void pack_weights_kernel(
    const float* __restrict__ W1, const float* __restrict__ W2,
    const float* __restrict__ W3,
    _Float16* __restrict__ w1h, _Float16* __restrict__ w2h,
    _Float16* __restrict__ w3h, unsigned char* __restrict__ acc0, int acc_n)
{
    int t = threadIdx.x;
    for (int i = t; i < 3 * HID;   i += 256) w1h[i] = (_Float16)W1[i];
    for (int i = t; i < HID * HID; i += 256) w2h[i] = (_Float16)W2[i];
    for (int i = t; i < HID;       i += 256) w3h[i] = (_Float16)W3[i];
    for (int i = t; i < acc_n;     i += 256) acc0[i] = 1;
}

// --- dense MLP evaluation: one wave = 16 grid points per tile, WMMA GEMMs ---
// Grid-stride over tiles so the weight-fragment setup is loop-invariant.
__global__ __launch_bounds__(256) void mlp_eval_kernel(
    const _Float16* __restrict__ w1h, const _Float16* __restrict__ w2h,
    const _Float16* __restrict__ w3h,
    const float* __restrict__ b1, const float* __restrict__ b2,
    const float* __restrict__ b3,
    float* __restrict__ occ, int r, int stride, int npts, int ntiles)
{
    __shared__ __align__(16) _Float16 Hlds[WAVES_PER_BLOCK][16 * HID];

    const int lane  = threadIdx.x & 31;
    const int wid   = threadIdx.x >> 5;
    const int half_ = lane >> 4;   // 0: lanes 0-15, 1: lanes 16-31
    const int lrow  = lane & 15;

    // ---- loop-invariant B fragments (32x16 f16):
    //      lane = column N (lrow), element e -> K = half*16 + e (B1), half*8 + e + (e >= 8 ? 8 : 0) (B2/B3)
    v16h B1[4];
    for (int c = 0; c < 4; ++c) {
        v16h b;
        int n = c * 16 + lrow;
        for (int e = 0; e < 16; ++e) {
            int k = half_ * 16 + e;
            b[e] = (k < 3) ? w1h[k * HID + n] : (_Float16)0.0f;
        }
        B1[c] = b;
    }
    v16h B2[2][4];
    for (int kc = 0; kc < 2; ++kc)
        for (int c = 0; c < 4; ++c) {
            v16h b;
            int n = c * 16 + lrow;
            for (int e = 0; e < 16; ++e) {
                int k = kc * 32 + (e < 8 ? half_ * 8 + e : 16 + half_ * 8 + (e - 8));
                b[e] = w2h[k * HID + n];
            }
            B2[kc][c] = b;
        }
    v16h B3[2];
    for (int kc = 0; kc < 2; ++kc) {
        v16h b;
        for (int e = 0; e < 16; ++e) {
            int k = kc * 32 + (e < 8 ? half_ * 8 + e : 16 + half_ * 8 + (e - 8));
            b[e] = (lrow == 0) ? w3h[k] : (_Float16)0.0f;
        }
        B3[kc] = b;
    }
    float b1v[4], b2v[4];
    for (int c = 0; c < 4; ++c) {
        b1v[c] = b1[c * 16 + lrow];
        b2v[c] = b2[c * 16 + lrow];
    }
    const float b3v = b3[0];

    const float sc = 2.0f / 257.0f;   // coords2D scaling, RES[-1] = 257
    const int wave0   = blockIdx.x * WAVES_PER_BLOCK + wid;
    const int nwaves  = gridDim.x * WAVES_PER_BLOCK;
    _Float16* Hw = &Hlds[wid][0];

    for (int tile = wave0; tile < ntiles; tile += nwaves) {
        const int base = tile * 16;

        // ---- A1 (16x32 f16): lanes<16 carry K=0..7 -> (x,y,z,0,...) -------
        v16h A1;
        for (int e = 0; e < 16; ++e) A1[e] = (_Float16)0.0f;
        if (half_ == 0) {
            int g = base + lrow;
            if (g >= npts) g = npts - 1;
            int w = g % r;
            int t = g / r;
            int h = t % r;
            int d = t / r;
            A1[0] = (_Float16)(((float)(w * stride) + 0.5f) * sc - 1.0f);
            A1[1] = (_Float16)(((float)(h * stride) + 0.5f) * sc - 1.0f);
            A1[2] = (_Float16)(((float)(d * stride) + 0.5f) * sc - 1.0f);
        }

        // ---- layer 1: (16x3) @ (3x64), bias seeded into C, ReLU -> LDS ----
        for (int c = 0; c < 4; ++c) {
            v8f acc;
            for (int i = 0; i < 8; ++i) acc[i] = b1v[c];
            acc = __builtin_amdgcn_wmma_f32_16x16x32_f16(false, A1, false, B1[c],
                                                         (short)0, acc, false, false);
            for (int i = 0; i < 8; ++i) {
                float v = acc[i];
                v = v > 0.0f ? v : 0.0f;
                Hw[(i + 8 * half_) * HID + c * 16 + lrow] = (_Float16)v;
            }
        }

        // ---- re-stripe H (C layout) -> A fragments via LDS ----------------
        v16h A2[2];
        for (int f = 0; f < 2; ++f) {
            int k0 = f * 32 + half_ * 8;
            half8 lo = *(const half8*)&Hw[lrow * HID + k0];
            half8 hi = *(const half8*)&Hw[lrow * HID + k0 + 16];
            v16h a;
            for (int e = 0; e < 8; ++e) { a[e] = lo[e]; a[8 + e] = hi[e]; }
            A2[f] = a;
        }

        // ---- layer 2: (16x64) @ (64x64), chained K accumulation -----------
        for (int c = 0; c < 4; ++c) {
            v8f acc;
            for (int i = 0; i < 8; ++i) acc[i] = b2v[c];
            acc = __builtin_amdgcn_wmma_f32_16x16x32_f16(false, A2[0], false, B2[0][c],
                                                         (short)0, acc, false, false);
            acc = __builtin_amdgcn_wmma_f32_16x16x32_f16(false, A2[1], false, B2[1][c],
                                                         (short)0, acc, false, false);
            for (int i = 0; i < 8; ++i) {
                float v = acc[i];
                v = v > 0.0f ? v : 0.0f;
                Hw[(i + 8 * half_) * HID + c * 16 + lrow] = (_Float16)v;
            }
        }

        v16h A3[2];
        for (int f = 0; f < 2; ++f) {
            int k0 = f * 32 + half_ * 8;
            half8 lo = *(const half8*)&Hw[lrow * HID + k0];
            half8 hi = *(const half8*)&Hw[lrow * HID + k0 + 16];
            v16h a;
            for (int e = 0; e < 8; ++e) { a[e] = lo[e]; a[8 + e] = hi[e]; }
            A3[f] = a;
        }

        // ---- layer 3: (16x64) @ (64x1 padded to 16), sigmoid --------------
        v8f acc3;
        for (int i = 0; i < 8; ++i) acc3[i] = b3v;
        acc3 = __builtin_amdgcn_wmma_f32_16x16x32_f16(false, A3[0], false, B3[0],
                                                      (short)0, acc3, false, false);
        acc3 = __builtin_amdgcn_wmma_f32_16x16x32_f16(false, A3[1], false, B3[1],
                                                      (short)0, acc3, false, false);

        // column N=0 lives in lanes 0 (M=0..7) and 16 (M=8..15)
        if (lrow == 0) {
            for (int i = 0; i < 8; ++i) {
                int g = base + half_ * 8 + i;
                if (g < npts) occ[g] = 1.0f / (1.0f + __expf(-acc3[i]));
            }
        }
    }
}

// --- trilinear 2x-1 upsample + raw boundary mask ----------------------------
__global__ __launch_bounds__(256) void upsample_kernel(
    const float* __restrict__ src, float* __restrict__ dst,
    unsigned char* __restrict__ raw, int p, int r)
{
    long long n   = (long long)r * r * r;
    long long idx = (long long)blockIdx.x * blockDim.x + threadIdx.x;
    if (idx >= n) return;
    int w = (int)(idx % r);
    long long t = idx / r;
    int h = (int)(t % r);
    int d = (int)(t / r);
    int pw = w >> 1, ph = h >> 1, pd = d >> 1;
    int ow = w & 1, oh = h & 1, od = d & 1;
    float s = 0.0f;
    int cnt = 0, nc = 0;
    for (int dz = 0; dz <= od; ++dz)
        for (int dy = 0; dy <= oh; ++dy)
            for (int dx = 0; dx <= ow; ++dx) {
                float v = src[((long long)(pd + dz) * p + (ph + dy)) * p + (pw + dx)];
                s += v;
                cnt += (v > 0.5f) ? 1 : 0;   // balance_value
                ++nc;
            }
    dst[idx] = s / (float)nc;
    raw[idx] = (cnt > 0 && cnt < nc) ? 1 : 0;   // strictly between 0 and 1
}

// --- box dilation (radius R), accum exclusion, masked scatter of occ_new ----
__global__ __launch_bounds__(256) void dilate_select_kernel(
    float* __restrict__ occ, const float* __restrict__ occ_new,
    const unsigned char* __restrict__ raw, const unsigned char* __restrict__ acc_prev,
    unsigned char* __restrict__ acc_cur, int r, int R)
{
    long long n   = (long long)r * r * r;
    long long idx = (long long)blockIdx.x * blockDim.x + threadIdx.x;
    if (idx >= n) return;
    int w = (int)(idx % r);
    long long t = idx / r;
    int h = (int)(t % r);
    int d = (int)(t / r);

    bool m = false;
    for (int dz = -R; dz <= R && !m; ++dz) {
        int z = d + dz;
        if ((unsigned)z >= (unsigned)r) continue;
        for (int dy = -R; dy <= R && !m; ++dy) {
            int y = h + dy;
            if ((unsigned)y >= (unsigned)r) continue;
            long long rowb = ((long long)z * r + y) * r;
            for (int dx = -R; dx <= R; ++dx) {
                int x = w + dx;
                if ((unsigned)x >= (unsigned)r) continue;
                if (raw[rowb + x]) { m = true; break; }
            }
        }
    }
    bool even = (((w | h | d) & 1) == 0);
    int p = (r + 1) >> 1;
    bool ap = even &&
        (acc_prev[((long long)(d >> 1) * p + (h >> 1)) * p + (w >> 1)] != 0);
    bool sel = m && !ap;
    if (sel) occ[idx] = occ_new[idx];
    acc_cur[idx] = (ap || sel) ? 1 : 0;
}

// --- final pure trilinear upsample 129 -> 257 (coalesced 67.9 MB store) -----
__global__ __launch_bounds__(256) void final_up_kernel(
    const float* __restrict__ src, float* __restrict__ dst, int p, int r)
{
    long long n   = (long long)r * r * r;
    long long idx = (long long)blockIdx.x * blockDim.x + threadIdx.x;
    if (idx >= n) return;
    int w = (int)(idx % r);
    long long t = idx / r;
    int h = (int)(t % r);
    int d = (int)(t / r);
    int pw = w >> 1, ph = h >> 1, pd = d >> 1;
    int ow = w & 1, oh = h & 1, od = d & 1;
    float s = 0.0f;
    int nc = 0;
    for (int dz = 0; dz <= od; ++dz)
        for (int dy = 0; dy <= oh; ++dy)
            for (int dx = 0; dx <= ow; ++dx) {
                s += src[((long long)(pd + dz) * p + (ph + dy)) * p + (pw + dx)];
                ++nc;
            }
    dst[idx] = s / (float)nc;
}

// ---------------------------------------------------------------------------

static inline int ceil_div_i(long long a, long long b) { return (int)((a + b - 1) / b); }

extern "C" void kernel_launch(void* const* d_in, const int* in_sizes, int n_in,
                              void* d_out, int out_size, void* d_ws, size_t ws_size,
                              hipStream_t stream)
{
    (void)in_sizes; (void)n_in; (void)out_size; (void)ws_size;

    const float* W1 = (const float*)d_in[0];   // (3,64)
    const float* b1 = (const float*)d_in[1];   // (64,)
    const float* W2 = (const float*)d_in[2];   // (64,64)
    const float* b2 = (const float*)d_in[3];   // (64,)
    const float* W3 = (const float*)d_in[4];   // (64,1)
    const float* b3 = (const float*)d_in[5];   // (1,)
    float* out = (float*)d_out;                // (257^3,)

    const int R17 = 17, R33 = 33, R65 = 65, R129 = 129, R257 = 257;
    const long long N17  = (long long)R17 * R17 * R17;     // 4913
    const long long N33  = (long long)R33 * R33 * R33;     // 35937
    const long long N65  = (long long)R65 * R65 * R65;     // 274625
    const long long N129 = (long long)R129 * R129 * R129;  // 2146689
    const long long N257 = (long long)R257 * R257 * R257;  // 16974593

    // workspace carve-out (~23 MB total)
    char* ws = (char*)d_ws;
    size_t off = 0;
    auto take = [&](size_t b) { size_t o = off; off += (b + 255) & ~(size_t)255; return o; };
    _Float16* w1h = (_Float16*)(ws + take(3 * HID * sizeof(_Float16)));
    _Float16* w2h = (_Float16*)(ws + take((size_t)HID * HID * sizeof(_Float16)));
    _Float16* w3h = (_Float16*)(ws + take(HID * sizeof(_Float16)));
    float* occ17  = (float*)(ws + take((size_t)N17  * sizeof(float)));
    float* occ33  = (float*)(ws + take((size_t)N33  * sizeof(float)));
    float* occ65  = (float*)(ws + take((size_t)N65  * sizeof(float)));
    float* occ129 = (float*)(ws + take((size_t)N129 * sizeof(float)));
    float* occ_new = (float*)(ws + take((size_t)N129 * sizeof(float)));  // shared
    unsigned char* raw  = (unsigned char*)(ws + take((size_t)N129));     // shared
    unsigned char* accA = (unsigned char*)(ws + take((size_t)N129));
    unsigned char* accB = (unsigned char*)(ws + take((size_t)N129));

    // weights -> f16, accum(level0) = all ones
    pack_weights_kernel<<<1, 256, 0, stream>>>(W1, W2, W3, w1h, w2h, w3h,
                                               accA, (int)N17);

    const int MAX_BLOCKS = 2048;   // persistent-wave cap: 16k waves
    auto mlp_launch = [&](float* dst, int r, int stride, long long npts) {
        int tiles  = (int)((npts + 15) / 16);
        int blocks = ceil_div_i(tiles, WAVES_PER_BLOCK);
        if (blocks > MAX_BLOCKS) blocks = MAX_BLOCKS;
        mlp_eval_kernel<<<blocks, 256, 0, stream>>>(
            w1h, w2h, w3h, b1, b2, b3, dst, r, stride, (int)npts, tiles);
    };

    // level 0: 17^3, stride 16
    mlp_launch(occ17, R17, 16, N17);

    // level 1: 33^3, stride 8, kernel 9 -> radius 4
    mlp_launch(occ_new, R33, 8, N33);
    upsample_kernel<<<ceil_div_i(N33, 256), 256, 0, stream>>>(occ17, occ33, raw, R17, R33);
    dilate_select_kernel<<<ceil_div_i(N33, 256), 256, 0, stream>>>(
        occ33, occ_new, raw, accA, accB, R33, 4);

    // level 2: 65^3, stride 4, kernel 7 -> radius 3
    mlp_launch(occ_new, R65, 4, N65);
    upsample_kernel<<<ceil_div_i(N65, 256), 256, 0, stream>>>(occ33, occ65, raw, R33, R65);
    dilate_select_kernel<<<ceil_div_i(N65, 256), 256, 0, stream>>>(
        occ65, occ_new, raw, accB, accA, R65, 3);

    // level 3: 129^3, stride 2, kernel 3 -> radius 1
    mlp_launch(occ_new, R129, 2, N129);
    upsample_kernel<<<ceil_div_i(N129, 256), 256, 0, stream>>>(occ65, occ129, raw, R65, R129);
    dilate_select_kernel<<<ceil_div_i(N129, 256), 256, 0, stream>>>(
        occ129, occ_new, raw, accA, accB, R129, 1);

    // level 4: pure trilinear upsample to 257^3
    final_up_kernel<<<ceil_div_i(N257, 256), 256, 0, stream>>>(occ129, out, R129, R257);
}
